// ComplementaryAttention_75625784148413
// MI455X (gfx1250) — compile-verified, matrix operand order checked by reading, not run on hardware
//
#include <hip/hip_runtime.h>
#include <hip/hip_bf16.h>

typedef __attribute__((ext_vector_type(16))) __bf16 bf16x16;
typedef __attribute__((ext_vector_type(2)))  __bf16 bf16x2;
typedef __attribute__((ext_vector_type(8)))  float  f32x8;

union FragB16 { bf16x16 v; unsigned u[8]; unsigned short s[16]; };

// native RNE f32->bf16 (v_cvt_*bf16_f32)
__device__ __forceinline__ unsigned short f2bf(float f) {
  __bf16 b = (__bf16)f;
  return __builtin_bit_cast(unsigned short, b);
}
// pack two converts as a native bf16 2-vector -> single v_cvt_pk_bf16_f32
__device__ __forceinline__ unsigned packbf(float a, float b) {
  bf16x2 p;
  p.x = (__bf16)a;
  p.y = (__bf16)b;
  return __builtin_bit_cast(unsigned, p);
}

// packed-uint index for fragment dword j (K pairs), given lane-half h2
__device__ __forceinline__ int k2map(int j, int h2) {
  return (j < 4) ? (4*h2 + j) : (8 + 4*h2 + (j - 4));
}

// ---------------------------------------------------------------------------
// K1: q/k/v = x @ W + b   (M=4096, K=512, N=512), out bf16 head-major (B,H,T,64)
// ---------------------------------------------------------------------------
__global__ __launch_bounds__(256)
void qkv_gemm_kernel(const float* __restrict__ x,
                     const float* __restrict__ Wq, const float* __restrict__ bq,
                     const float* __restrict__ Wk, const float* __restrict__ bk,
                     const float* __restrict__ Wv, const float* __restrict__ bv,
                     unsigned short* __restrict__ qo,
                     unsigned short* __restrict__ ko,
                     unsigned short* __restrict__ vo) {
  const int D = 512, N = 512;
  const float* W; const float* bias; unsigned short* out;
  if (blockIdx.z == 0)      { W = Wq; bias = bq; out = qo; }
  else if (blockIdx.z == 1) { W = Wk; bias = bk; out = ko; }
  else                      { W = Wv; bias = bv; out = vo; }

  __shared__ unsigned aP[128 * 16];   // A tile 128x32 bf16 packed pairs along K
  __shared__ unsigned bP[64 * 16];    // B tile 32x64 bf16, transposed+packed

  const int t = threadIdx.x;
  const int wave = t >> 5, lane = t & 31;
  const int h2 = lane >> 4, lm = lane & 15;
  const int m0 = blockIdx.x * 128;
  const int n0 = blockIdx.y * 64;

  f32x8 acc[4] = {};

  for (int k0 = 0; k0 < D; k0 += 32) {
#pragma unroll
    for (int j = 0; j < 8; ++j) {                // stage x 128x32 (2048 pairs)
      int p = t + 256 * j;
      int row = p >> 4, k2 = p & 15;
      const float2* xp = (const float2*)(x + (size_t)(m0 + row) * D + k0);
      float2 vv = xp[k2];
      aP[row * 16 + k2] = packbf(vv.x, vv.y);
    }
#pragma unroll
    for (int j = 0; j < 4; ++j) {                // stage W 32x64 (1024 pairs)
      int p = t + 256 * j;
      int n = p & 63, k2 = p >> 6;
      float w0 = W[(size_t)(k0 + 2 * k2) * N + n0 + n];
      float w1 = W[(size_t)(k0 + 2 * k2 + 1) * N + n0 + n];
      bP[n * 16 + k2] = packbf(w0, w1);
    }
    __syncthreads();

    FragB16 aF;
    const int mrow = wave * 16 + lm;
#pragma unroll
    for (int j = 0; j < 8; ++j) aF.u[j] = aP[mrow * 16 + k2map(j, h2)];

#pragma unroll
    for (int nt = 0; nt < 4; ++nt) {
      FragB16 bF;
      const int nc = nt * 16 + lm;
#pragma unroll
      for (int j = 0; j < 8; ++j) bF.u[j] = bP[nc * 16 + k2map(j, h2)];
      acc[nt] = __builtin_amdgcn_wmma_f32_16x16x32_bf16(
          false, aF.v, false, bF.v, (short)0, acc[nt], false, false);
    }
    __syncthreads();
  }

#pragma unroll
  for (int nt = 0; nt < 4; ++nt) {
#pragma unroll
    for (int i = 0; i < 8; ++i) {
      int row = m0 + wave * 16 + i + 8 * h2;
      int col = n0 + nt * 16 + lm;
      float val = acc[nt][i] + bias[col];
      int b = row >> 10, tt = row & 1023;
      int h = col >> 6,  d  = col & 63;
      out[((size_t)(b * 8 + h) * 1024 + tt) * 64 + d] = f2bf(val);
    }
  }
}

// ---------------------------------------------------------------------------
// K2: S = (q @ k^T) * scale, per (b,h): 1024x1024, K=64
// ---------------------------------------------------------------------------
__global__ __launch_bounds__(256)
void scores_kernel(const unsigned short* __restrict__ q,
                   const unsigned short* __restrict__ k,
                   float* __restrict__ S) {
  const int bh = blockIdx.z;
  const unsigned* qg = (const unsigned*)(q + (size_t)bh * 1024 * 64);
  const unsigned* kg = (const unsigned*)(k + (size_t)bh * 1024 * 64);
  float* Sg = S + (size_t)bh * 1024 * 1024;

  const int t = threadIdx.x;
  const int wave = t >> 5, lane = t & 31;
  const int h2 = lane >> 4, lm = lane & 15;
  const int q0 = blockIdx.x * 128;
  const int c0 = blockIdx.y * 64;

  f32x8 acc[4] = {};
#pragma unroll
  for (int d0 = 0; d0 < 64; d0 += 32) {
    FragB16 aF;
    const int m = q0 + wave * 16 + lm;
#pragma unroll
    for (int j = 0; j < 8; ++j) aF.u[j] = qg[m * 32 + (d0 >> 1) + k2map(j, h2)];
#pragma unroll
    for (int nt = 0; nt < 4; ++nt) {
      FragB16 bF;
      const int n = c0 + nt * 16 + lm;
#pragma unroll
      for (int j = 0; j < 8; ++j) bF.u[j] = kg[n * 32 + (d0 >> 1) + k2map(j, h2)];
      acc[nt] = __builtin_amdgcn_wmma_f32_16x16x32_bf16(
          false, aF.v, false, bF.v, (short)0, acc[nt], false, false);
    }
  }
  const float scale = 0.125f;   // 64^-0.5
#pragma unroll
  for (int nt = 0; nt < 4; ++nt)
#pragma unroll
    for (int i = 0; i < 8; ++i) {
      int row = q0 + wave * 16 + i + 8 * h2;
      int col = c0 + nt * 16 + lm;
      Sg[(size_t)row * 1024 + col] = acc[nt][i] * scale;
    }
}

// ---------------------------------------------------------------------------
// K3: in-place row softmax over 1024 elements (one block per row)
// ---------------------------------------------------------------------------
__global__ __launch_bounds__(256)
void softmax_kernel(float* __restrict__ S) {
  __shared__ float red[256];
  float* p = S + (size_t)blockIdx.x * 1024;
  const int t = threadIdx.x;
  float v[4];
  float mx = -3.4e38f;
#pragma unroll
  for (int j = 0; j < 4; ++j) { v[j] = p[t + 256 * j]; mx = fmaxf(mx, v[j]); }
  red[t] = mx; __syncthreads();
  for (int s = 128; s > 0; s >>= 1) { if (t < s) red[t] = fmaxf(red[t], red[t + s]); __syncthreads(); }
  mx = red[0]; __syncthreads();
  float sum = 0.f;
#pragma unroll
  for (int j = 0; j < 4; ++j) { v[j] = __expf(v[j] - mx); sum += v[j]; }
  red[t] = sum; __syncthreads();
  for (int s = 128; s > 0; s >>= 1) { if (t < s) red[t] += red[t + s]; __syncthreads(); }
  float inv = 1.0f / red[0];
#pragma unroll
  for (int j = 0; j < 4; ++j) p[t + 256 * j] = v[j] * inv;
}

// ---------------------------------------------------------------------------
// K4: fused gate: R_f = sigmoid(sum_h A_t*w_h + b); masks; re-softmax both
//     branches; bf16 outputs. One block per (b, q-row), 8 heads in registers.
// ---------------------------------------------------------------------------
__global__ __launch_bounds__(256)
void gate_kernel(const float* __restrict__ At,
                 const float* __restrict__ conv_w, const float* __restrict__ conv_b,
                 unsigned short* __restrict__ Af, unsigned short* __restrict__ Ar) {
  __shared__ float red[256];
  __shared__ float cw[9];
  __shared__ float sums[16];
  const int t = threadIdx.x;
  if (t < 8) cw[t] = conv_w[t];
  if (t == 8) cw[8] = conv_b[0];
  __syncthreads();

  const int bq = blockIdx.x;                 // b*1024 + qrow
  const int b = bq >> 10, qrow = bq & 1023;

  float a[8][4];
#pragma unroll
  for (int h = 0; h < 8; ++h) {
    const float* p = At + ((size_t)(b * 8 + h) * 1024 + qrow) * 1024;
#pragma unroll
    for (int j = 0; j < 4; ++j) a[h][j] = p[t + 256 * j];
  }
  float mf[4], mr[4];
#pragma unroll
  for (int j = 0; j < 4; ++j) {
    float r = cw[8];
#pragma unroll
    for (int h = 0; h < 8; ++h) r += a[h][j] * cw[h];
    float sig = 1.0f / (1.0f + __expf(-r));
    mf[j] = (sig <= 0.5f) ? 1.0f : 0.0f;
    mr[j] = ((1.0f - sig) <= 0.5f) ? 1.0f : 0.0f;
  }
  float part[16];
#pragma unroll
  for (int h = 0; h < 8; ++h) {
    float sf = 0.f, sr = 0.f;
#pragma unroll
    for (int j = 0; j < 4; ++j) {
      sf += __expf(a[h][j] * mf[j]);
      sr += __expf(a[h][j] * mr[j]);
    }
    part[h] = sf; part[8 + h] = sr;
  }
  for (int i = 0; i < 16; ++i) {
    red[t] = part[i]; __syncthreads();
    for (int s = 128; s > 0; s >>= 1) { if (t < s) red[t] += red[t + s]; __syncthreads(); }
    if (t == 0) sums[i] = red[0];
    __syncthreads();
  }
#pragma unroll
  for (int h = 0; h < 8; ++h) {
    const size_t base = ((size_t)(b * 8 + h) * 1024 + qrow) * 1024;
    const float invf = 1.0f / sums[h], invr = 1.0f / sums[8 + h];
#pragma unroll
    for (int j = 0; j < 4; ++j) {
      int kk = t + 256 * j;
      Af[base + kk] = f2bf(__expf(a[h][j] * mf[j]) * invf);
      Ar[base + kk] = f2bf(__expf(a[h][j] * mr[j]) * invr);
    }
  }
}

// ---------------------------------------------------------------------------
// K5: T = A @ v per (b,h): [1024x1024]@[1024x64]; whole V tile staged into LDS
//     with CDNA5 async global->LDS B128 copies (ASYNCcnt-tracked).
// ---------------------------------------------------------------------------
__global__ __launch_bounds__(256)
void av_kernel(const unsigned short* __restrict__ A,
               const unsigned short* __restrict__ V,
               unsigned short* __restrict__ out) {
  extern __shared__ unsigned short vs[];          // 1024*64 bf16 = 128 KB
  const int bh = blockIdx.z;
  const int b = bh >> 3, hh = bh & 7;
  const unsigned* Ag = (const unsigned*)(A + (size_t)bh * 1024 * 1024);
  const unsigned short* Vg = V + (size_t)bh * 1024 * 64;

  const int t = threadIdx.x;
  {
    // async copy: each lane moves 32 x 16B; per-wave ASYNCcnt peaks at 32 (<63)
    const unsigned ldsbase = (unsigned)(size_t)vs;
#pragma unroll
    for (int j = 0; j < 32; ++j) {
      int idx = (t + 256 * j) * 8;                        // bf16 elements
      unsigned long long ga = (unsigned long long)(size_t)(Vg + idx);
      unsigned la = ldsbase + (unsigned)idx * 2;
      asm volatile("global_load_async_to_lds_b128 %0, %1, off"
                   :: "v"(la), "v"(ga) : "memory");
    }
#if __has_builtin(__builtin_amdgcn_s_wait_asynccnt)
    __builtin_amdgcn_s_wait_asynccnt(0);
#else
    asm volatile("s_wait_asynccnt 0" ::: "memory");
#endif
  }
  __syncthreads();

  const int wave = t >> 5, lane = t & 31;
  const int h2 = lane >> 4, lm = lane & 15;
  const int q0 = blockIdx.x * 128;

  f32x8 acc[4] = {};
  for (int kk0 = 0; kk0 < 1024; kk0 += 32) {
    FragB16 aF;
    const int m = q0 + wave * 16 + lm;
#pragma unroll
    for (int j = 0; j < 8; ++j) aF.u[j] = Ag[m * 512 + (kk0 >> 1) + k2map(j, h2)];
#pragma unroll
    for (int nt = 0; nt < 4; ++nt) {
      FragB16 bF;
      const int n = nt * 16 + lm;
#pragma unroll
      for (int e = 0; e < 16; ++e) {
        int K = kk0 + ((e < 8) ? (8 * h2 + e) : (16 + 8 * h2 + (e - 8)));
        bF.s[e] = vs[K * 64 + n];
      }
      acc[nt] = __builtin_amdgcn_wmma_f32_16x16x32_bf16(
          false, aF.v, false, bF.v, (short)0, acc[nt], false, false);
    }
  }
#pragma unroll
  for (int nt = 0; nt < 4; ++nt)
#pragma unroll
    for (int i = 0; i < 8; ++i) {
      int qrow = q0 + wave * 16 + i + 8 * h2;
      int d = nt * 16 + lm;
      out[((size_t)(b * 1024 + qrow)) * 512 + hh * 64 + d] = f2bf(acc[nt][i]);
    }
}

// ---------------------------------------------------------------------------
// K6: out = A(bf16, 4096x512) @ W(f32, 512x512) + bias -> f32
// ---------------------------------------------------------------------------
__global__ __launch_bounds__(256)
void out_gemm_kernel(const unsigned short* __restrict__ A,
                     const float* __restrict__ W,
                     const float* __restrict__ bias,
                     float* __restrict__ out) {
  __shared__ unsigned bP[64 * 16];
  const unsigned* Ag = (const unsigned*)A;
  const int t = threadIdx.x;
  const int wave = t >> 5, lane = t & 31;
  const int h2 = lane >> 4, lm = lane & 15;
  const int m0 = blockIdx.x * 128, n0 = blockIdx.y * 64;
  f32x8 acc[4] = {};
  for (int k0 = 0; k0 < 512; k0 += 32) {
#pragma unroll
    for (int j = 0; j < 4; ++j) {                // stage W 32x64 (1024 pairs)
      int p = t + 256 * j;
      int n = p & 63, k2 = p >> 6;
      float w0 = W[(size_t)(k0 + 2 * k2) * 512 + n0 + n];
      float w1 = W[(size_t)(k0 + 2 * k2 + 1) * 512 + n0 + n];
      bP[n * 16 + k2] = packbf(w0, w1);
    }
    __syncthreads();
    FragB16 aF;
    const int m = m0 + wave * 16 + lm;
#pragma unroll
    for (int j = 0; j < 8; ++j) aF.u[j] = Ag[m * 256 + (k0 >> 1) + k2map(j, h2)];
#pragma unroll
    for (int nt = 0; nt < 4; ++nt) {
      FragB16 bF;
      const int nc = nt * 16 + lm;
#pragma unroll
      for (int j = 0; j < 8; ++j) bF.u[j] = bP[nc * 16 + k2map(j, h2)];
      acc[nt] = __builtin_amdgcn_wmma_f32_16x16x32_bf16(
          false, aF.v, false, bF.v, (short)0, acc[nt], false, false);
    }
    __syncthreads();
  }
#pragma unroll
  for (int nt = 0; nt < 4; ++nt)
#pragma unroll
    for (int i = 0; i < 8; ++i) {
      int row = m0 + wave * 16 + i + 8 * h2;
      int col = n0 + nt * 16 + lm;
      out[(size_t)row * 512 + col] = acc[nt][i] + bias[col];
    }
}

// ---------------------------------------------------------------------------
// K7: pairwise softmax gate: w0 = sigmoid(sigmoid(Gf)-sigmoid(Gr)); blend
// ---------------------------------------------------------------------------
__global__ __launch_bounds__(256)
void combine_kernel(const float* __restrict__ Ff, const float* __restrict__ Gfl,
                    const float* __restrict__ Fr, const float* __restrict__ Grl,
                    float* __restrict__ out) {
  int i = blockIdx.x * 256 + threadIdx.x;
  float gf = 1.0f / (1.0f + __expf(-Gfl[i]));
  float gr = 1.0f / (1.0f + __expf(-Grl[i]));
  float w0 = 1.0f / (1.0f + __expf(gr - gf));   // softmax over {gf, gr}
  out[i] = Ff[i] * w0 + Fr[i] * (1.0f - w0);
}

extern "C" void kernel_launch(void* const* d_in, const int* in_sizes, int n_in,
                              void* d_out, int out_size, void* d_ws, size_t ws_size,
                              hipStream_t stream) {
  (void)in_sizes; (void)n_in; (void)out_size; (void)ws_size;
  const float* x   = (const float*)d_in[0];
  const float* Wq  = (const float*)d_in[1];
  const float* bq  = (const float*)d_in[2];
  const float* Wk  = (const float*)d_in[3];
  const float* bk  = (const float*)d_in[4];
  const float* Wv  = (const float*)d_in[5];
  const float* bv  = (const float*)d_in[6];
  const float* cw  = (const float*)d_in[7];
  const float* cb  = (const float*)d_in[8];
  const float* Wfh = (const float*)d_in[9];
  const float* bfh = (const float*)d_in[10];
  const float* Wfg = (const float*)d_in[11];
  const float* bfg = (const float*)d_in[12];
  const float* Wrh = (const float*)d_in[13];
  const float* brh = (const float*)d_in[14];
  const float* Wrg = (const float*)d_in[15];
  const float* brg = (const float*)d_in[16];

  char* ws = (char*)d_ws;
  unsigned short* q  = (unsigned short*)(ws);                    // 4 MB
  unsigned short* k  = (unsigned short*)(ws + (4ull   << 20));   // 4 MB
  unsigned short* v  = (unsigned short*)(ws + (8ull   << 20));   // 4 MB
  float*          S  = (float*)        (ws + (12ull  << 20));    // 128 MB (A_t in place)
  unsigned short* Af = (unsigned short*)(ws + (140ull << 20));   // 64 MB
  unsigned short* Ar = (unsigned short*)(ws + (204ull << 20));   // 64 MB
  unsigned short* Tf = (unsigned short*)(ws + (268ull << 20));   // 4 MB
  unsigned short* Tr = (unsigned short*)(ws + (272ull << 20));   // 4 MB
  float*          Ff = (float*)        (ws + (276ull << 20));    // 8 MB
  float*          Gf = (float*)        (ws + (284ull << 20));    // 8 MB
  float*          Fr = (float*)        (ws + (292ull << 20));    // 8 MB
  float*          Gr = (float*)        (ws + (300ull << 20));    // 8 MB  (end: 308 MB)

  qkv_gemm_kernel<<<dim3(32, 8, 3), 256, 0, stream>>>(x, Wq, bq, Wk, bk, Wv, bv, q, k, v);
  scores_kernel<<<dim3(8, 16, 32), 256, 0, stream>>>(q, k, S);
  softmax_kernel<<<dim3(32768), 256, 0, stream>>>(S);
  gate_kernel<<<dim3(4096), 256, 0, stream>>>(S, cw, cb, Af, Ar);
  av_kernel<<<dim3(8, 1, 32), 256, 131072, stream>>>(Af, v, Tf);
  av_kernel<<<dim3(8, 1, 32), 256, 131072, stream>>>(Ar, v, Tr);
  out_gemm_kernel<<<dim3(32, 8), 256, 0, stream>>>(Tf, Wfh, bfh, Ff);
  out_gemm_kernel<<<dim3(32, 8), 256, 0, stream>>>(Tf, Wfg, bfg, Gf);
  out_gemm_kernel<<<dim3(32, 8), 256, 0, stream>>>(Tr, Wrh, brh, Fr);
  out_gemm_kernel<<<dim3(32, 8), 256, 0, stream>>>(Tr, Wrg, brg, Gr);
  combine_kernel<<<dim3(8192), 256, 0, stream>>>(Ff, Gf, Fr, Gr, (float*)d_out);
}
